// AttnGNNLayer_6657199309416
// MI455X (gfx1250) — matrix operand order repaired and checked by reading, not compiled, not run on hardware
//
#include <hip/hip_runtime.h>
#include <hip/hip_bf16.h>
#include <math.h>

// ---------------------------------------------------------------------------
// CDNA5 (gfx1250) implementation of the AttnGNN layer.
// All heavy GEMMs (Gram/kNN, EdgeConv, 512-ch expansion) use
// v_wmma_f32_16x16x32_f16 with f32 accumulation. BatchNorm statistics are
// global over the batch, so each bn-dependent stage is two-pass
// (stats pass -> finalize -> apply pass), recomputing the cheap GEMM.
// ---------------------------------------------------------------------------

typedef __attribute__((ext_vector_type(16))) _Float16 v16h;
typedef __attribute__((ext_vector_type(8)))  float    v8f;

union FragH {
    v16h h;
    unsigned int u[8];
    _Float16 e[16];
};

#define NPTS 256          // points per group
#define NGRP 512          // B*M groups
#define KNN  8

// -------------------------------------------------------------------------
// Build x0 = concat(xyz^T, feats^T) : (G, 32, 256)
// -------------------------------------------------------------------------
__global__ void __launch_bounds__(256)
build_x_kernel(const float* __restrict__ xyz, const float* __restrict__ feats,
               float* __restrict__ x0)
{
    int g = blockIdx.x, n = threadIdx.x;
    const float* pz = xyz   + ((size_t)g * NPTS + n) * 3;
    const float* pf = feats + ((size_t)g * NPTS + n) * 29;
    float* xo = x0 + (size_t)g * 32 * NPTS;
    #pragma unroll
    for (int j = 0; j < 3; ++j)  xo[j * NPTS + n]       = pz[j];
    #pragma unroll
    for (int j = 0; j < 29; ++j) xo[(3 + j) * NPTS + n] = pf[j];
}

// -------------------------------------------------------------------------
// Gram matrix + top-8 kNN per group.  One block per group, 8 waves.
// S = X^T X computed with v_wmma_f32_16x16x32_f16; top-8 of
// pd = 2*S - xx_n - xx_m maintained per row in registers.
// -------------------------------------------------------------------------
template<int C>
__global__ void __launch_bounds__(256)
gram_knn_kernel(const float* __restrict__ xin, int gstride,
                int* __restrict__ idxout)
{
    __shared__ __align__(16) _Float16 xt[NPTS * C];   // [n][c]
    __shared__ float xx[NPTS];
    __shared__ float tileS[8][16][17];                // per-wave strip

    int g = blockIdx.x, tid = threadIdx.x;
    const float* xg = xin + (size_t)g * gstride * NPTS;

    { // stage x (f16) + norms (f32)
        int n = tid;
        float s = 0.f;
        for (int c = 0; c < C; ++c) {
            float v = xg[c * NPTS + n];
            s += v * v;
            xt[n * C + c] = (_Float16)v;
        }
        xx[n] = s;
    }
    __syncthreads();

    int wave = tid >> 5, lane = tid & 31, l15 = lane & 15;
    bool hi = lane >= 16;
    const unsigned* xtu = (const unsigned*)xt;

    for (int pass = 0; pass < 2; ++pass) {
        int nt = wave + pass * 8;          // n-tile 0..15
        int row = nt * 16 + l15;

        FragH A[C / 32];
        #pragma unroll
        for (int ks = 0; ks < C / 32; ++ks) {
            int kb = ks * 32;
            #pragma unroll
            for (int v = 0; v < 8; ++v) {
                int cp = kb + ((v < 4) ? 2 * v : 8 + 2 * v) + (hi ? 8 : 0);
                A[ks].u[v] = xtu[(row * C + cp) >> 1];
            }
        }

        float tv[8]; int ti[8];
        #pragma unroll
        for (int q = 0; q < 8; ++q) { tv[q] = -3.0e38f; ti[q] = 0; }

        for (int mt = 0; mt < 16; ++mt) {
            v8f acc = {};
            #pragma unroll
            for (int ks = 0; ks < C / 32; ++ks) {
                int kb = ks * 32;
                FragH Bf;
                int colm = mt * 16 + l15;
                #pragma unroll
                for (int v = 0; v < 8; ++v) {
                    int cp = kb + ((v < 4) ? 2 * v : 8 + 2 * v) + (hi ? 8 : 0);
                    Bf.u[v] = xtu[(colm * C + cp) >> 1];
                }
                acc = __builtin_amdgcn_wmma_f32_16x16x32_f16(
                        false, A[ks].h, false, Bf.h, (short)0, acc, false, false);
            }
            // spill tile to the wave-private LDS strip (same-wave LDS is in-order)
            #pragma unroll
            for (int v = 0; v < 8; ++v)
                tileS[wave][v + (hi ? 8 : 0)][l15] = acc[v];
            asm volatile("" ::: "memory");

            if (lane < 16) {                    // merge 16 candidates into top-8
                int r = l15;
                float xn = xx[nt * 16 + r];
                for (int mm = 0; mm < 16; ++mm) {
                    int m = mt * 16 + mm;
                    float v = 2.f * tileS[wave][r][mm] - xn - xx[m];
                    if (v > tv[7]) {
                        int p = 7;
                        while (p > 0 && v > tv[p - 1]) {
                            tv[p] = tv[p - 1]; ti[p] = ti[p - 1]; --p;
                        }
                        tv[p] = v; ti[p] = m;
                    }
                }
            }
            asm volatile("" ::: "memory");
        }
        if (lane < 16) {
            int n = nt * 16 + l15;
            #pragma unroll
            for (int q = 0; q < 8; ++q)
                idxout[((size_t)g * NPTS + n) * KNN + q] = ti[q];
        }
    }
}

// -------------------------------------------------------------------------
// EdgeConv: h[o, n, k] = W(o, :) . [x_nb - x_n ; x_n]   (O = 64, K2 = 2C)
// MODE 0: accumulate per-channel sum / sumsq.
// MODE 1: bn + relu + max over k -> xout[g, off+o, n].
// -------------------------------------------------------------------------
template<int C, int MODE>
__global__ void __launch_bounds__(256)
edgeconv_kernel(const float* __restrict__ xin, int gstride,
                const int* __restrict__ idxin,
                const float* __restrict__ W,          // (64, 2C)
                const float* __restrict__ gamma, const float* __restrict__ beta,
                const float* __restrict__ mu, const float* __restrict__ rs,
                float* __restrict__ ssum, float* __restrict__ ssq,
                float* __restrict__ xout, int out_off)
{
    constexpr int C2 = 2 * C;
    __shared__ __align__(16) _Float16 xt[NPTS * C];
    __shared__ __align__(16) _Float16 wt[64 * C2];
    __shared__ int nbIdx[NPTS * KNN];

    int g = blockIdx.x, tid = threadIdx.x;
    const float* xg = xin + (size_t)g * gstride * NPTS;
    {
        int n = tid;
        for (int c = 0; c < C; ++c) xt[n * C + c] = (_Float16)xg[c * NPTS + n];
    }
    for (int i = tid; i < 64 * C2; i += 256) wt[i] = (_Float16)W[i];
    for (int i = tid; i < NPTS * KNN; i += 256)
        nbIdx[i] = idxin[(size_t)g * NPTS * KNN + i];
    __syncthreads();

    int wave = tid >> 5, lane = tid & 31, l15 = lane & 15;
    bool hi = lane >= 16;
    const unsigned* xtu = (const unsigned*)xt;
    const unsigned* wtu = (const unsigned*)wt;

    for (int ot = 0; ot < 4; ++ot) {
        int orow = ot * 16 + l15;
        FragH A[C2 / 32];
        #pragma unroll
        for (int ks = 0; ks < C2 / 32; ++ks) {
            int kb = ks * 32;
            #pragma unroll
            for (int v = 0; v < 8; ++v) {
                int cp = kb + ((v < 4) ? 2 * v : 8 + 2 * v) + (hi ? 8 : 0);
                A[ks].u[v] = wtu[(orow * C2 + cp) >> 1];
            }
        }

        float s8[8], q8[8], m8[8], r8[8], g8[8], b8[8];
        #pragma unroll
        for (int v = 0; v < 8; ++v) { s8[v] = 0.f; q8[v] = 0.f; }
        if (MODE == 1) {
            #pragma unroll
            for (int v = 0; v < 8; ++v) {
                int o = ot * 16 + v + (hi ? 8 : 0);
                m8[v] = mu[o]; r8[v] = rs[o]; g8[v] = gamma[o]; b8[v] = beta[o];
            }
        }

        for (int i = 0; i < 16; ++i) {
            int jt = wave * 16 + i;            // column tile 0..127
            int col = jt * 16 + l15;           // j = n*8 + k
            int n  = col >> 3;
            int nb = nbIdx[col];
            v8f acc = {};
            #pragma unroll
            for (int ks = 0; ks < C2 / 32; ++ks) {
                int kb = ks * 32;
                FragH Bf;
                #pragma unroll
                for (int v = 0; v < 8; ++v) {
                    int cc = kb + ((v < 4) ? 2 * v : 8 + 2 * v) + (hi ? 8 : 0);
                    if (cc < C) {              // uniform per ks (C multiple of 32)
                        Bf.e[2 * v]     = xt[nb * C + cc]     - xt[n * C + cc];
                        Bf.e[2 * v + 1] = xt[nb * C + cc + 1] - xt[n * C + cc + 1];
                    } else {
                        Bf.u[v] = xtu[(n * C + (cc - C)) >> 1];
                    }
                }
                acc = __builtin_amdgcn_wmma_f32_16x16x32_f16(
                        false, A[ks].h, false, Bf.h, (short)0, acc, false, false);
            }
            if (MODE == 0) {
                #pragma unroll
                for (int v = 0; v < 8; ++v) {
                    float h = acc[v];
                    s8[v] += h; q8[v] += h * h;
                }
            } else {
                #pragma unroll
                for (int v = 0; v < 8; ++v) {
                    float h = (acc[v] - m8[v]) * r8[v] * g8[v] + b8[v];
                    h = fmaxf(h, 0.f);
                    h = fmaxf(h, __shfl_xor(h, 1));   // max over k within 8 lanes
                    h = fmaxf(h, __shfl_xor(h, 2));
                    h = fmaxf(h, __shfl_xor(h, 4));
                    if ((lane & 7) == 0) {
                        int o  = ot * 16 + v + (hi ? 8 : 0);
                        int nn = jt * 2 + (l15 >> 3);
                        xout[((size_t)g * 128 + out_off + o) * NPTS + nn] = h;
                    }
                }
            }
        }
        if (MODE == 0) {
            #pragma unroll
            for (int v = 0; v < 8; ++v) {
                int o = ot * 16 + v + (hi ? 8 : 0);
                atomicAdd(&ssum[o], s8[v]);
                atomicAdd(&ssq[o],  q8[v]);
            }
        }
    }
}

// -------------------------------------------------------------------------
// finalize bn stats: mu = s/cnt, rs = rsqrt(var + eps)
// -------------------------------------------------------------------------
__global__ void finalize_bn_kernel(const float* __restrict__ s,
                                   const float* __restrict__ q,
                                   float* __restrict__ mu, float* __restrict__ rs,
                                   int C, float invcnt)
{
    int i = blockIdx.x * blockDim.x + threadIdx.x;
    if (i < C) {
        float m   = s[i] * invcnt;
        float var = q[i] * invcnt - m * m;
        mu[i] = m;
        rs[i] = rsqrtf(var + 1e-5f);
    }
}

// -------------------------------------------------------------------------
// Sigmoid gate: c1 = relu(bn(wc1 . x)); c2 = wc2 . c1 + bc2; x *= sigmoid(c2)
// MODE 0: stats of c1.  MODE 1: apply gate in place.
// -------------------------------------------------------------------------
template<int MODE>
__global__ void __launch_bounds__(256)
gate_kernel(float* __restrict__ xcat,
            const float* __restrict__ wc1, const float* __restrict__ gc1,
            const float* __restrict__ bc1, const float* __restrict__ wc2,
            const float* __restrict__ bc2,
            const float* __restrict__ mu, const float* __restrict__ rs,
            float* __restrict__ ssum, float* __restrict__ ssq)
{
    __shared__ float xl[16][NPTS];
    __shared__ float red_s[64], red_q[64];
    int g = blockIdx.x, n = threadIdx.x;

    float acc[64];
    #pragma unroll
    for (int o = 0; o < 64; ++o) acc[o] = 0.f;

    for (int cb = 0; cb < 128; cb += 16) {
        __syncthreads();
        #pragma unroll
        for (int c = 0; c < 16; ++c)
            xl[c][n] = xcat[((size_t)g * 128 + cb + c) * NPTS + n];
        __syncthreads();
        for (int o = 0; o < 64; ++o) {
            float a = acc[o];
            #pragma unroll
            for (int c = 0; c < 16; ++c)
                a += wc1[o * 128 + cb + c] * xl[c][n];
            acc[o] = a;
        }
    }

    if (MODE == 0) {
        if (n < 64) { red_s[n] = 0.f; red_q[n] = 0.f; }
        __syncthreads();
        for (int o = 0; o < 64; ++o) {
            atomicAdd(&red_s[o], acc[o]);
            atomicAdd(&red_q[o], acc[o] * acc[o]);
        }
        __syncthreads();
        if (n < 64) { atomicAdd(&ssum[n], red_s[n]); atomicAdd(&ssq[n], red_q[n]); }
    } else {
        float c1n[64];
        #pragma unroll
        for (int o = 0; o < 64; ++o)
            c1n[o] = fmaxf((acc[o] - mu[o]) * rs[o] * gc1[o] + bc1[o], 0.f);
        for (int cc = 0; cc < 128; ++cc) {
            float t = bc2[cc];
            for (int o = 0; o < 64; ++o) t += wc2[cc * 64 + o] * c1n[o];
            float sgm = 1.f / (1.f + __expf(-t));
            size_t off = ((size_t)g * 128 + cc) * NPTS + n;
            xcat[off] = xcat[off] * sgm;
        }
    }
}

// -------------------------------------------------------------------------
// Expansion: e = relu(bn(wex . x)) (512 ch) then max over n.
// WMMA GEMM (512x128)x(128x256) per group.
// MODE 0: stats of pre-activation.  MODE 1: bn+relu+max -> emax[g, o].
// -------------------------------------------------------------------------
template<int MODE>
__global__ void __launch_bounds__(256)
ex_kernel(const float* __restrict__ xg, const float* __restrict__ wex,
          const float* __restrict__ gamma, const float* __restrict__ beta,
          const float* __restrict__ mu, const float* __restrict__ rs,
          float* __restrict__ ssum, float* __restrict__ ssq,
          float* __restrict__ emax)
{
    __shared__ __align__(16) _Float16 xt[NPTS * 128];   // 64 KB
    int g = blockIdx.x, tid = threadIdx.x;
    {
        int n = tid;
        for (int c = 0; c < 128; ++c)
            xt[n * 128 + c] = (_Float16)xg[((size_t)g * 128 + c) * NPTS + n];
    }
    __syncthreads();

    int wave = tid >> 5, lane = tid & 31, l15 = lane & 15;
    bool hi = lane >= 16;
    const unsigned* xtu = (const unsigned*)xt;

    for (int oi = 0; oi < 4; ++oi) {
        int ot = wave * 4 + oi;          // 0..31 (512 channels)
        int orow = ot * 16 + l15;
        FragH A[4];
        #pragma unroll
        for (int ks = 0; ks < 4; ++ks) {
            int kb = ks * 32;
            #pragma unroll
            for (int v = 0; v < 8; ++v) {
                int cp = kb + ((v < 4) ? 2 * v : 8 + 2 * v) + (hi ? 8 : 0);
                A[ks].e[2 * v]     = (_Float16)wex[orow * 128 + cp];
                A[ks].e[2 * v + 1] = (_Float16)wex[orow * 128 + cp + 1];
            }
        }

        float s8[8], q8[8], mx8[8], m8[8], r8[8], g8[8], b8[8];
        #pragma unroll
        for (int v = 0; v < 8; ++v) { s8[v] = 0.f; q8[v] = 0.f; mx8[v] = -3.0e38f; }
        if (MODE == 1) {
            #pragma unroll
            for (int v = 0; v < 8; ++v) {
                int o = ot * 16 + v + (hi ? 8 : 0);
                m8[v] = mu[o]; r8[v] = rs[o]; g8[v] = gamma[o]; b8[v] = beta[o];
            }
        }

        for (int nt = 0; nt < 16; ++nt) {
            int coln = nt * 16 + l15;
            v8f acc = {};
            #pragma unroll
            for (int ks = 0; ks < 4; ++ks) {
                int kb = ks * 32;
                FragH Bf;
                #pragma unroll
                for (int v = 0; v < 8; ++v) {
                    int cc = kb + ((v < 4) ? 2 * v : 8 + 2 * v) + (hi ? 8 : 0);
                    Bf.u[v] = xtu[(coln * 128 + cc) >> 1];
                }
                acc = __builtin_amdgcn_wmma_f32_16x16x32_f16(
                        false, A[ks].h, false, Bf.h, (short)0, acc, false, false);
            }
            if (MODE == 0) {
                #pragma unroll
                for (int v = 0; v < 8; ++v) { float h = acc[v]; s8[v] += h; q8[v] += h * h; }
            } else {
                #pragma unroll
                for (int v = 0; v < 8; ++v) {
                    float h = fmaxf((acc[v] - m8[v]) * r8[v] * g8[v] + b8[v], 0.f);
                    mx8[v] = fmaxf(mx8[v], h);
                }
            }
        }
        if (MODE == 0) {
            #pragma unroll
            for (int v = 0; v < 8; ++v) {
                int o = ot * 16 + v + (hi ? 8 : 0);
                atomicAdd(&ssum[o], s8[v]);
                atomicAdd(&ssq[o],  q8[v]);
            }
        } else {
            #pragma unroll
            for (int v = 0; v < 8; ++v) {
                float h = mx8[v];
                h = fmaxf(h, __shfl_xor(h, 1));   // max over 16 n-lanes
                h = fmaxf(h, __shfl_xor(h, 2));
                h = fmaxf(h, __shfl_xor(h, 4));
                h = fmaxf(h, __shfl_xor(h, 8));
                if (l15 == 0) {
                    int o = ot * 16 + v + (hi ? 8 : 0);
                    emax[(size_t)g * 512 + o] = h;
                }
            }
        }
    }
}

// ------------------------- small tail kernels ----------------------------
__global__ void __launch_bounds__(256)
tail_wr_kernel(const float* __restrict__ emax, const float* __restrict__ wr,
               float* __restrict__ y, float* __restrict__ ssum, float* __restrict__ ssq)
{
    __shared__ float col[512];
    int cidx = blockIdx.x;               // b*256 + m
    int i = threadIdx.x;
    for (int o = i; o < 512; o += 256) col[o] = emax[(size_t)cidx * 512 + o];
    __syncthreads();
    float a = 0.f;
    for (int o = 0; o < 512; ++o) a += wr[i * 512 + o] * col[o];
    int b = cidx >> 8, m = cidx & 255;
    y[((size_t)b * 256 + i) * 256 + m] = a;
    atomicAdd(&ssum[i], a); atomicAdd(&ssq[i], a * a);
}

__global__ void __launch_bounds__(256)
tail_bn1_kernel(const float* __restrict__ y, const float* __restrict__ gr,
                const float* __restrict__ br, const float* __restrict__ mu,
                const float* __restrict__ rs, float* __restrict__ xd,
                float* __restrict__ ssum, float* __restrict__ ssq)
{
    int cidx = blockIdx.x; int i = threadIdx.x;
    int b = cidx >> 8, m = cidx & 255;
    size_t off = ((size_t)b * 256 + i) * 256 + m;
    float v = y[off];
    v = fmaxf((v - mu[i]) * rs[i] * gr[i] + br[i], 0.f) * 2.0f;  // relu then x+x
    xd[off] = v;
    atomicAdd(&ssum[i], v); atomicAdd(&ssq[i], v * v);
}

__global__ void __launch_bounds__(256)
tail_mlp_kernel(const float* __restrict__ xd,
                const float* __restrict__ mu2, const float* __restrict__ rs2,
                const float* __restrict__ gsn1, const float* __restrict__ bsn1,
                const float* __restrict__ ws1, const float* __restrict__ bs1,
                const float* __restrict__ ws2, const float* __restrict__ bs2,
                float* __restrict__ z, float* __restrict__ ssum, float* __restrict__ ssq)
{
    __shared__ float xs[256][16];
    __shared__ float hh[256][16];
    int colblk = blockIdx.x;             // 32 blocks * 16 cols
    int i = threadIdx.x;
    for (int c2 = 0; c2 < 16; ++c2) {
        int cidx = colblk * 16 + c2; int b = cidx >> 8, m = cidx & 255;
        float v = xd[((size_t)b * 256 + i) * 256 + m];
        xs[i][c2] = (v - mu2[i]) * rs2[i] * gsn1[i] + bsn1[i];
    }
    __syncthreads();
    for (int c2 = 0; c2 < 16; ++c2) {
        float a = bs1[i];
        for (int c = 0; c < 256; ++c) a += ws1[i * 256 + c] * xs[c][c2];
        hh[i][c2] = fmaxf(a, 0.f);
    }
    __syncthreads();
    float ls = 0.f, lq = 0.f;
    for (int c2 = 0; c2 < 16; ++c2) {
        float a = bs2[i];
        for (int c = 0; c < 256; ++c) a += ws2[i * 256 + c] * hh[c][c2];
        float zz = xs[i][c2] + a;
        int cidx = colblk * 16 + c2; int b = cidx >> 8, m = cidx & 255;
        z[((size_t)b * 256 + i) * 256 + m] = zz;
        ls += zz; lq += zz * zz;
    }
    atomicAdd(&ssum[i], ls); atomicAdd(&ssq[i], lq);
}

__global__ void __launch_bounds__(256)
tail_out_kernel(const float* __restrict__ z, const float* __restrict__ mu3,
                const float* __restrict__ rs3, const float* __restrict__ gsn2,
                const float* __restrict__ bsn2, float* __restrict__ out)
{
    int cidx = blockIdx.x; int i = threadIdx.x;
    int b = cidx >> 8, m = cidx & 255;
    size_t off = ((size_t)b * 256 + i) * 256 + m;
    out[off] = (z[off] - mu3[i]) * rs3[i] * gsn2[i] + bsn2[i];
}

// -------------------------------------------------------------------------
extern "C" void kernel_launch(void* const* d_in, const int* in_sizes, int n_in,
                              void* d_out, int out_size, void* d_ws, size_t ws_size,
                              hipStream_t stream)
{
    (void)in_sizes; (void)n_in; (void)out_size; (void)ws_size;

    const float* xyz  = (const float*)d_in[0];
    const float* feats= (const float*)d_in[1];
    const float* we1  = (const float*)d_in[2];
    const float* g1   = (const float*)d_in[3];
    const float* b1   = (const float*)d_in[4];
    const float* we2  = (const float*)d_in[5];
    const float* g2   = (const float*)d_in[6];
    const float* b2   = (const float*)d_in[7];
    const float* wc1  = (const float*)d_in[8];
    const float* gc1  = (const float*)d_in[9];
    const float* bc1  = (const float*)d_in[10];
    const float* wc2  = (const float*)d_in[11];
    const float* bc2  = (const float*)d_in[12];
    const float* wex  = (const float*)d_in[13];
    const float* gex  = (const float*)d_in[14];
    const float* bex  = (const float*)d_in[15];
    const float* wr   = (const float*)d_in[16];
    const float* gr   = (const float*)d_in[17];
    const float* br   = (const float*)d_in[18];
    const float* ws1  = (const float*)d_in[19];
    const float* bs1  = (const float*)d_in[20];
    const float* ws2  = (const float*)d_in[21];
    const float* bs2  = (const float*)d_in[22];
    const float* gsn1 = (const float*)d_in[23];
    const float* bsn1 = (const float*)d_in[24];
    const float* gsn2 = (const float*)d_in[25];
    const float* bsn2 = (const float*)d_in[26];
    float* out = (float*)d_out;

    // ---- workspace layout (floats) ----
    float* w     = (float*)d_ws;
    float* x0    = w;                        // 512*32*256
    float* xcat  = x0   + 4194304;           // 512*128*256
    float* emaxb = xcat + 16777216;          // 512*512
    float* ybuf  = emaxb + 262144;           // 2*256*256
    float* xdbuf = ybuf  + 131072;
    float* zbuf  = xdbuf + 131072;
    float* ssum  = zbuf  + 131072;           // 512
    float* ssq   = ssum  + 512;
    float* mub   = ssq   + 512;
    float* rsb   = mub   + 512;
    int*   idxb  = (int*)(rsb + 512);        // 512*256*8

    dim3 blk(256);
    const float invEC   = 1.0f / 1048576.0f; // G*N*K
    const float invGN   = 1.0f / 131072.0f;  // G*N
    const float invBM   = 1.0f / 512.0f;     // B*M

    build_x_kernel<<<NGRP, blk, 0, stream>>>(xyz, feats, x0);

    // ---- EdgeConv layer 1 (C=32 -> 64) ----
    gram_knn_kernel<32><<<NGRP, blk, 0, stream>>>(x0, 32, idxb);
    hipMemsetAsync(ssum, 0, 1024 * sizeof(float), stream);
    edgeconv_kernel<32, 0><<<NGRP, blk, 0, stream>>>(x0, 32, idxb, we1, g1, b1,
                                                     mub, rsb, ssum, ssq, xcat, 0);
    finalize_bn_kernel<<<1, 64, 0, stream>>>(ssum, ssq, mub, rsb, 64, invEC);
    edgeconv_kernel<32, 1><<<NGRP, blk, 0, stream>>>(x0, 32, idxb, we1, g1, b1,
                                                     mub, rsb, ssum, ssq, xcat, 0);

    // ---- EdgeConv layer 2 (C=64 -> 64, input = xcat ch 0..63) ----
    gram_knn_kernel<64><<<NGRP, blk, 0, stream>>>(xcat, 128, idxb);
    hipMemsetAsync(ssum, 0, 1024 * sizeof(float), stream);
    edgeconv_kernel<64, 0><<<NGRP, blk, 0, stream>>>(xcat, 128, idxb, we2, g2, b2,
                                                     mub, rsb, ssum, ssq, xcat, 64);
    finalize_bn_kernel<<<1, 64, 0, stream>>>(ssum, ssq, mub, rsb, 64, invEC);
    edgeconv_kernel<64, 1><<<NGRP, blk, 0, stream>>>(xcat, 128, idxb, we2, g2, b2,
                                                     mub, rsb, ssum, ssq, xcat, 64);

    // ---- sigmoid gate ----
    hipMemsetAsync(ssum, 0, 1024 * sizeof(float), stream);
    gate_kernel<0><<<NGRP, blk, 0, stream>>>(xcat, wc1, gc1, bc1, wc2, bc2,
                                             mub, rsb, ssum, ssq);
    finalize_bn_kernel<<<1, 64, 0, stream>>>(ssum, ssq, mub, rsb, 64, invGN);
    gate_kernel<1><<<NGRP, blk, 0, stream>>>(xcat, wc1, gc1, bc1, wc2, bc2,
                                             mub, rsb, ssum, ssq);

    // ---- 512-ch expansion + max-pool over n ----
    hipMemsetAsync(ssum, 0, 1024 * sizeof(float), stream);
    ex_kernel<0><<<NGRP, blk, 0, stream>>>(xcat, wex, gex, bex, mub, rsb,
                                           ssum, ssq, emaxb);
    finalize_bn_kernel<<<2, 256, 0, stream>>>(ssum, ssq, mub, rsb, 512, invGN);
    ex_kernel<1><<<NGRP, blk, 0, stream>>>(xcat, wex, gex, bex, mub, rsb,
                                           ssum, ssq, emaxb);

    // ---- tail ----
    hipMemsetAsync(ssum, 0, 1024 * sizeof(float), stream);
    tail_wr_kernel<<<512, blk, 0, stream>>>(emaxb, wr, ybuf, ssum, ssq);
    finalize_bn_kernel<<<1, 256, 0, stream>>>(ssum, ssq, mub, rsb, 256, invBM);

    hipMemsetAsync(ssum, 0, 1024 * sizeof(float), stream);
    tail_bn1_kernel<<<512, blk, 0, stream>>>(ybuf, gr, br, mub, rsb, xdbuf, ssum, ssq);
    finalize_bn_kernel<<<1, 256, 0, stream>>>(ssum, ssq, mub, rsb, 256, invBM);

    hipMemsetAsync(ssum, 0, 1024 * sizeof(float), stream);
    tail_mlp_kernel<<<32, blk, 0, stream>>>(xdbuf, mub, rsb, gsn1, bsn1,
                                            ws1, bs1, ws2, bs2, zbuf, ssum, ssq);
    finalize_bn_kernel<<<1, 256, 0, stream>>>(ssum, ssq, mub, rsb, 256, invBM);

    tail_out_kernel<<<512, blk, 0, stream>>>(zbuf, mub, rsb, gsn2, bsn2, out);
}
